// TGATEncoder_33586644255097
// MI455X (gfx1250) — matrix operand order repaired and checked by reading, not compiled, not run on hardware
//
#include <hip/hip_runtime.h>
#include <math.h>

typedef __attribute__((ext_vector_type(16))) _Float16 v16h;
typedef __attribute__((ext_vector_type(8)))  float    v8f;

#define N_NODES  50000
#define N_EDGES  800000
#define IN_CH    128
#define OUT_CH   128
#define TIME_DIM 64
#define FAN      (IN_CH + TIME_DIM)   // 192
#define EPB      64                   // edges per block
#define KSTEPS   (FAN / 32)           // 6 WMMA K-steps

// ---------- workspace layout (bytes, 64B aligned) ----------
#define WS_WT     0                       // wTn: FAN*OUT_CH f16 (N-major) = 49,152
#define WS_ALPHA  49152                   // alphaBuf: N_EDGES f32 = 3,200,000
#define WS_EX     (WS_ALPHA + 3200000)    // exBuf: N_EDGES f32
#define WS_SEGMAX (WS_EX + 3200000)       // segMaxKey: N_NODES u32
#define WS_DENOM  (WS_SEGMAX + 200000)    // denom: N_NODES f32
#define WS_HBUF   (WS_DENOM + 200000)     // hbuf: N_EDGES*OUT_CH f16 = 204,800,000

// Monotone float<->u32 key map so unsigned atomicMax == float max.
__device__ __forceinline__ unsigned floatToKey(float f) {
    unsigned b = __float_as_uint(f);
    return (b & 0x80000000u) ? ~b : (b | 0x80000000u);
}
__device__ __forceinline__ float keyToFloat(unsigned k) {
    unsigned b = (k & 0x80000000u) ? (k ^ 0x80000000u) : ~k;
    return __uint_as_float(b);
}

// Branchless tanh: 1 - 2/(exp(2z)+1); saturates to +/-1 for large |z|.
__device__ __forceinline__ float fast_tanh(float z) {
    float t = __expf(2.0f * z);
    return fmaf(-2.0f, __builtin_amdgcn_rcpf(t + 1.0f), 1.0f);
}

// ---------------- kernel 1: init out / segmax / denom ----------------
__global__ void tgat_init(float* __restrict__ out, unsigned* __restrict__ segMaxKey,
                          float* __restrict__ denom) {
    int i = blockIdx.x * blockDim.x + threadIdx.x;
    if (i < N_NODES * OUT_CH) out[i] = 0.0f;
    if (i < N_NODES) { segMaxKey[i] = 0u; denom[i] = 0.0f; }
}

// ------------- kernel 2: lin_w [OUT][FAN] f32 -> f16 (same N-major layout) -------------
__global__ void tgat_convw(const float* __restrict__ lin_w, _Float16* __restrict__ wTn) {
    int i = blockIdx.x * blockDim.x + threadIdx.x;
    if (i < FAN * OUT_CH) wTn[i] = (_Float16)lin_w[i];
}

// ---------------- kernel 3: edge GEMM via WMMA ----------------
// Block: 256 threads (8 waves), 64 edges. Wave w owns N-slice [16w, 16w+16).
// x-row gather DMA'd into LDS via per-lane async-to-LDS; B fragments loaded as
// contiguous v16h from N-major f16 weights (L2 resident).
__global__ void __launch_bounds__(256) tgat_gemm(
    const float* __restrict__ x, const int* __restrict__ src,
    const int* __restrict__ dst, const float* __restrict__ edge_t,
    const float* __restrict__ freqs, const _Float16* __restrict__ wTn,
    const float* __restrict__ lin_b, const float* __restrict__ attn,
    _Float16* __restrict__ hbuf, float* __restrict__ alphaBuf,
    unsigned* __restrict__ segMaxKey)
{
    __shared__ float    sXf[EPB][IN_CH];     // 32 KB: async-DMA gather staging (f32)
    __shared__ _Float16 sA[EPB][FAN + 8];    // 25.6 KB: WMMA A tile (f16), padded
    __shared__ float    sAlpha[EPB];

    const int tid = threadIdx.x;
    const int e0  = blockIdx.x * EPB;

    if (tid < EPB) sAlpha[tid] = 0.0f;

    // ---- async DMA: gather x[src[e]] rows (16B per lane per sweep) into sXf ----
    {
        const unsigned ldsBase = (unsigned)(uintptr_t)&sXf[0][0];
#pragma unroll
        for (int s = 0; s < (EPB * IN_CH * 4) / (256 * 16); ++s) {   // 8 sweeps
            int b = s * 256 + tid;              // 16B chunk id, 32 chunks per edge row
            int e = b >> 5;
            int c = b & 31;
            const float* gaddr = x + (long)src[e0 + e] * IN_CH + c * 4;
            unsigned laddr = ldsBase + (unsigned)b * 16u;
            asm volatile("global_load_async_to_lds_b128 %0, %1, off"
                         :: "v"(laddr), "v"(gaddr) : "memory");
        }
    }

    // ---- time encoding directly into the A tile (independent of the DMA) ----
    for (int i = tid; i < EPB * TIME_DIM; i += 256) {
        int e = i >> 6;
        int j = i & 63;
        float ang = edge_t[e0 + e] * freqs[j & 31] * 6.28318530717958647692f;
        sA[e][IN_CH + j] = (_Float16)((j < 32) ? __sinf(ang) : __cosf(ang));
    }

    asm volatile("s_wait_asynccnt 0" ::: "memory");   // this wave's DMAs landed
    __syncthreads();                                  // => all waves' DMAs landed

    // ---- convert gathered f32 rows to f16 in the A tile ----
    for (int i = tid; i < EPB * IN_CH; i += 256) {
        int e = i >> 7;
        int k = i & 127;
        sA[e][k] = (_Float16)sXf[e][k];
    }
    __syncthreads();

    const int lane = tid & 31;
    const int g    = lane >> 4;                // lane group
    const int nloc = lane & 15;
    const int ncol = (tid >> 5) * 16 + nloc;   // this lane's N column (0..127)

    // ---- B fragments: halves j -> K = kb*32 + 16g + j; contiguous in N-major wTn ----
    v16h bfrag[KSTEPS];
#pragma unroll
    for (int kb = 0; kb < KSTEPS; ++kb)
        bfrag[kb] = *(const v16h*)(wTn + ncol * FAN + kb * 32 + 16 * g);

    const float bias = lin_b[ncol];
    const float aw   = attn[ncol];

#pragma unroll
    for (int mt = 0; mt < 4; ++mt) {           // 4 M-tiles of 16 edges
        v8f acc = {};
#pragma unroll
        for (int kb = 0; kb < KSTEPS; ++kb) {
            // A fragment: M = lane&15; halves j -> K = 16g + j (matches B)
            v16h a;
#pragma unroll
            for (int j = 0; j < 16; ++j) {
                int kl = 16 * g + j;
                a[j] = sA[mt * 16 + nloc][kb * 32 + kl];
            }
            acc = __builtin_amdgcn_wmma_f32_16x16x32_f16(
                false, a, false, bfrag[kb], (short)0, acc, false, false);
        }
        // epilogue: bias + tanh, write h (f16), accumulate alpha in LDS
#pragma unroll
        for (int v = 0; v < 8; ++v) {
            int m = mt * 16 + v + 8 * g;       // edge row in C/D layout
            float hv = fast_tanh(acc[v] + bias);
            hbuf[(long)(e0 + m) * OUT_CH + ncol] = (_Float16)hv;
            atomicAdd(&sAlpha[m], hv * aw);    // ds_add_f32
        }
    }
    __syncthreads();

    if (tid < EPB) {
        float a = sAlpha[tid];
        alphaBuf[e0 + tid] = a;
        atomicMax(&segMaxKey[dst[e0 + tid]], floatToKey(a));
    }
}

// ---------------- kernel 4: ex = exp(alpha - segmax[dst]); denom += ex ----------------
__global__ void tgat_expdenom(const int* __restrict__ dst,
                              const float* __restrict__ alphaBuf,
                              const unsigned* __restrict__ segMaxKey,
                              float* __restrict__ exBuf, float* __restrict__ denom) {
    int e = blockIdx.x * blockDim.x + threadIdx.x;
    if (e >= N_EDGES) return;
    int d = dst[e];
    float m  = keyToFloat(segMaxKey[d]);
    float ex = __expf(alphaBuf[e] - m);
    exBuf[e] = ex;
    atomicAdd(&denom[d], ex);
}

// ---------------- kernel 5: out[dst] += h * ex/denom ----------------
__global__ void tgat_scatter(const int* __restrict__ dst,
                             const _Float16* __restrict__ hbuf,
                             const float* __restrict__ exBuf,
                             const float* __restrict__ denom,
                             float* __restrict__ out) {
    int i = blockIdx.x * blockDim.x + threadIdx.x;   // < N_EDGES*OUT_CH
    int e = i >> 7;
    int n = i & (OUT_CH - 1);
    if (e >= N_EDGES) return;
    int d = dst[e];
    float w = exBuf[e] / (denom[d] + 1e-16f);
    atomicAdd(&out[d * OUT_CH + n], (float)hbuf[i] * w);
}

extern "C" void kernel_launch(void* const* d_in, const int* in_sizes, int n_in,
                              void* d_out, int out_size, void* d_ws, size_t ws_size,
                              hipStream_t stream) {
    const float* x      = (const float*)d_in[0];
    const int*   eidx   = (const int*)d_in[1];        // [2, N_EDGES]
    const float* edge_t = (const float*)d_in[2];
    const float* freqs  = (const float*)d_in[3];
    const float* lin_w  = (const float*)d_in[4];      // [OUT_CH, FAN]
    const float* lin_b  = (const float*)d_in[5];
    const float* attn   = (const float*)d_in[6];
    float*       out    = (float*)d_out;

    const int* src = eidx;
    const int* dst = eidx + N_EDGES;

    char* ws = (char*)d_ws;
    _Float16* wTn       = (_Float16*)(ws + WS_WT);
    float*    alphaBuf  = (float*)(ws + WS_ALPHA);
    float*    exBuf     = (float*)(ws + WS_EX);
    unsigned* segMaxKey = (unsigned*)(ws + WS_SEGMAX);
    float*    denom     = (float*)(ws + WS_DENOM);
    _Float16* hbuf      = (_Float16*)(ws + WS_HBUF);

    tgat_init<<<(N_NODES * OUT_CH + 255) / 256, 256, 0, stream>>>(out, segMaxKey, denom);
    tgat_convw<<<(FAN * OUT_CH + 255) / 256, 256, 0, stream>>>(lin_w, wTn);
    tgat_gemm<<<N_EDGES / EPB, 256, 0, stream>>>(x, src, dst, edge_t, freqs, wTn,
                                                 lin_b, attn, hbuf, alphaBuf, segMaxKey);
    tgat_expdenom<<<(N_EDGES + 255) / 256, 256, 0, stream>>>(dst, alphaBuf, segMaxKey,
                                                             exBuf, denom);
    tgat_scatter<<<(N_EDGES * OUT_CH + 255) / 256, 256, 0, stream>>>(dst, hbuf, exBuf,
                                                                     denom, out);
}
